// RegionProposalNetwork_44882408243744
// MI455X (gfx1250) — matrix operand order repaired and checked by reading, not compiled, not run on hardware
//
#include <hip/hip_runtime.h>
#include <hip/hip_bf16.h>

// ---------------- problem constants ----------------
#define N_IMG   8
#define C_IN    512
#define C_OUT   512
#define PIX     4096          // 64*64
#define K_TOT   4608          // 512*9
#define M_TOT   32768         // 8*64*64
#define NANCH   36864         // 4096*9
#define POSTN   2000

// GEMM tiling for the 3x3 conv (implicit GEMM)
// K ordering: k = s*512 + ci, s = ky*3+kx  (chunk-constant spatial tap)
#define BM 64
#define BN 128
#define BK 64

typedef __attribute__((ext_vector_type(16))) _Float16 v16h;
typedef __attribute__((ext_vector_type(8)))  _Float16 v8h;
typedef __attribute__((ext_vector_type(8)))  float    v8f;
typedef __attribute__((ext_vector_type(4)))  int      v4i;

#if defined(__has_builtin)
#if __has_builtin(__builtin_amdgcn_global_load_async_to_lds_b128)
#define HAVE_ASYNC_LDS 1
#endif
#endif

__device__ __forceinline__ void wait_asynccnt0() {
#if defined(__has_builtin) && __has_builtin(__builtin_amdgcn_s_wait_asynccnt)
    __builtin_amdgcn_s_wait_asynccnt(0);
#else
    asm volatile("s_wait_asynccnt 0x0" ::: "memory");
#endif
}

// ---------------- K0: weights f32 -> f16, permuted to k = s*512 + ci ----------------
__global__ void k_cvt_w(const float* __restrict__ w, _Float16* __restrict__ wf, int n) {
    int i = blockIdx.x * 256 + threadIdx.x;      // output index
    if (i >= n) return;
    int o   = i / K_TOT;
    int rem = i - o * K_TOT;
    int s   = rem >> 9;                          // 0..8  (ky*3+kx)
    int ci  = rem & 511;
    wf[i] = (_Float16)w[o * K_TOT + ci * 9 + s];
}

// ---------------- K1: 3x3 conv + bias + ReLU via WMMA ----------------
__global__ __launch_bounds__(256) void k_conv3(
    const float* __restrict__ x, const _Float16* __restrict__ wf,
    const float* __restrict__ bias, float* __restrict__ h1) {

    __shared__ __align__(16) _Float16 As[BM][72];   // 144B rows -> 16B aligned frags
    __shared__ __align__(16) _Float16 Bs[BN][72];

    const int tid  = threadIdx.x;
    const int lane = tid & 31;
    const int wid  = tid >> 5;     // 0..7
    const int wm   = wid & 3;      // M sub-tile (16 rows each)
    const int wn   = wid >> 2;     // N sub-tile (64 cols each)

    const int mt = blockIdx.x & 511;   // 512 M tiles
    const int nt = blockIdx.x >> 9;    // 4   N tiles
    const int m_base = mt * BM;
    const int n_base = nt * BN;

    v8f acc[4] = {v8f{0.f}, v8f{0.f}, v8f{0.f}, v8f{0.f}};

    // A staging: thread -> (row 0..63, 16-wide ci segment)
    const int a_row  = tid >> 2;
    const int a_kseg = (tid & 3) * 16;
    const int m   = m_base + a_row;
    const int img = m >> 12;
    const int pix = m & 4095;
    const int oy  = pix >> 6;
    const int ox  = pix & 63;
    const float* ximg = x + (((long)img * C_IN) << 12);

    // B staging: thread -> (n row 0..127, 32-wide ci segment)
    const int  b_row  = tid >> 1;
    const int  b_kseg = (tid & 1) * 32;
    const _Float16* wrow = wf + (long)(n_base + b_row) * K_TOT + b_kseg;

    // fragment read coords (fixed per lane)
    const int arow = wm * 16 + (lane & 15);
    const int kb   = (lane >> 4) * 8;   // A: K halves base within 32-sub-chunk
    const int kbB  = (lane >> 4) * 8;   // B: K halves base within 32-sub-chunk

    for (int s_i = 0; s_i < 9; ++s_i) {
        const int ky = s_i / 3, kx = s_i - (s_i / 3) * 3;   // uniform scalars
        const int iy = oy + ky - 1;
        const int ix = ox + kx - 1;
        const bool inb = ((unsigned)iy < 64u) && ((unsigned)ix < 64u);
        const float* xpix = ximg + (iy << 6) + ix;          // + ci*4096 per element
        const _Float16* wseg = wrow + (long)s_i * 512;

        for (int c0 = 0; c0 < 512; c0 += BK) {
            // ---- stage B: async DMA global -> LDS (ASYNCcnt path) ----
            {
                const _Float16* src = wseg + c0;
#ifdef HAVE_ASYNC_LDS
                __attribute__((address_space(1))) v4i* gsrc =
                    (__attribute__((address_space(1))) v4i*)(src);
                __attribute__((address_space(3))) v4i* ldst =
                    (__attribute__((address_space(3))) v4i*)&Bs[b_row][b_kseg];
                __builtin_amdgcn_global_load_async_to_lds_b128(gsrc, ldst,  0, 0);
                __builtin_amdgcn_global_load_async_to_lds_b128(gsrc, ldst, 16, 0);
                __builtin_amdgcn_global_load_async_to_lds_b128(gsrc, ldst, 32, 0);
                __builtin_amdgcn_global_load_async_to_lds_b128(gsrc, ldst, 48, 0);
#else
                v8h p0 = *(const v8h*)(src);
                v8h p1 = *(const v8h*)(src + 8);
                v8h p2 = *(const v8h*)(src + 16);
                v8h p3 = *(const v8h*)(src + 24);
                *(v8h*)&Bs[b_row][b_kseg]      = p0;
                *(v8h*)&Bs[b_row][b_kseg + 8]  = p1;
                *(v8h*)&Bs[b_row][b_kseg + 16] = p2;
                *(v8h*)&Bs[b_row][b_kseg + 24] = p3;
#endif
                __builtin_prefetch(src + BK, 0, 1);   // global_prefetch_b8
            }

            // ---- stage A: 16 strided channel loads -> two b128 LDS stores ----
            v8h av0, av1;
            if (inb) {
                float f[16];
                const float* xp = xpix + ((long)(c0 + a_kseg) << 12);
                #pragma unroll
                for (int j = 0; j < 16; ++j) f[j] = xp[(long)j << 12];
                #pragma unroll
                for (int j = 0; j < 8; ++j) { av0[j] = (_Float16)f[j]; av1[j] = (_Float16)f[j + 8]; }
            } else {
                #pragma unroll
                for (int j = 0; j < 8; ++j) { av0[j] = (_Float16)0.f; av1[j] = (_Float16)0.f; }
            }
            *(v8h*)&As[a_row][a_kseg]     = av0;
            *(v8h*)&As[a_row][a_kseg + 8] = av1;

#ifdef HAVE_ASYNC_LDS
            wait_asynccnt0();     // drain async LDS writes before the barrier
#endif
            __syncthreads();

            // ---- 2 x (A-frag x 4 B-frags) WMMA ----
            #pragma unroll
            for (int kk = 0; kk < BK; kk += 32) {
                v8h alo = *(const v8h*)&As[arow][kk + kb];        // K = kb..kb+7
                v8h ahi = *(const v8h*)&As[arow][kk + kb + 16];   // K = kb+16..kb+23
                v16h a = __builtin_shufflevector(alo, ahi,
                         0,1,2,3,4,5,6,7,8,9,10,11,12,13,14,15);
                #pragma unroll
                for (int f = 0; f < 4; ++f) {
                    const int col = wn * 64 + f * 16 + (lane & 15);
                    v8h blo = *(const v8h*)&Bs[col][kk + kbB];
                    v8h bhi = *(const v8h*)&Bs[col][kk + kbB + 16];
                    v16h b = __builtin_shufflevector(blo, bhi,
                             0,1,2,3,4,5,6,7,8,9,10,11,12,13,14,15);
                    acc[f] = __builtin_amdgcn_wmma_f32_16x16x32_f16(
                                 false, a, false, b, (short)0, acc[f], false, false);
                }
            }
            __syncthreads();
        }
    }

    // ---- epilogue: bias + ReLU, store NCHW h1 ----
    const int mrow0 = wm * 16 + ((lane >> 4) * 8);
    #pragma unroll
    for (int f = 0; f < 4; ++f) {
        int c = n_base + wn * 64 + f * 16 + (lane & 15);
        float bv = bias[c];
        #pragma unroll
        for (int e = 0; e < 8; ++e) {
            int mg   = m_base + mrow0 + e;
            int img2 = mg >> 12;
            int pix2 = mg & 4095;
            float val = acc[f][e] + bv;
            val = val > 0.f ? val : 0.f;
            h1[(((long)img2 * C_OUT + c) << 12) + pix2] = val;
        }
    }
}

// ---------------- K2: 1x1 score/loc heads ----------------
__global__ void k_heads(const float* __restrict__ h1,
                        const float* __restrict__ score_w, const float* __restrict__ score_b,
                        const float* __restrict__ loc_w,   const float* __restrict__ loc_b,
                        float* __restrict__ out_scores, float* __restrict__ out_locs) {
    long t = (long)blockIdx.x * 256 + threadIdx.x;
    if (t >= (long)54 * M_TOT) return;
    int oc  = (int)(t >> 15);        // 0..53
    int m   = (int)(t & 32767);
    int img = m >> 12;
    int pix = m & 4095;
    const float* wrow  = (oc < 18) ? (score_w + (long)oc * 512)
                                   : (loc_w + (long)(oc - 18) * 512);
    const float* hbase = h1 + (((long)img * C_OUT) << 12) + pix;
    float acc = 0.f;
    #pragma unroll 4
    for (int c = 0; c < 512; c += 4) {
        float4 wv = *(const float4*)(wrow + c);
        float h0 = hbase[(long)(c + 0) << 12];
        float h1v = hbase[(long)(c + 1) << 12];
        float h2 = hbase[(long)(c + 2) << 12];
        float h3 = hbase[(long)(c + 3) << 12];
        acc += h0 * wv.x + h1v * wv.y + h2 * wv.z + h3 * wv.w;
    }
    if (oc < 18) {
        acc += score_b[oc];
        out_scores[(long)img * 73728 + (long)pix * 18 + oc] = acc;   // [n][site*18+c]
    } else {
        int c2 = oc - 18;
        acc += loc_b[c2];
        out_locs[(long)img * 147456 + (long)pix * 36 + c2] = acc;    // [n][site*36+c]
    }
}

// ---------------- K3: anchors + softmax fg + loc2bbox + clip + min-size ----------------
__global__ void k_decode(const float* __restrict__ out_scores, const float* __restrict__ out_locs,
                         const int* __restrict__ p_imgh, const int* __restrict__ p_imgw,
                         float* __restrict__ boxes, float* __restrict__ nscore,
                         float* __restrict__ anchor_out) {
    int t = blockIdx.x * 256 + threadIdx.x;
    if (t >= N_IMG * NANCH) return;
    int img  = t / NANCH;
    int ai   = t - img * NANCH;        // (y*64+x)*9 + a
    int a    = ai % 9;
    int site = ai / 9;
    int y = site >> 6, x = site & 63;

    const float ratios[3] = {0.5f, 1.f, 2.f};
    const float scales[3] = {8.f, 16.f, 32.f};
    float r = ratios[a / 3], s = scales[a % 3];
    float bh = 16.f * s * __builtin_sqrtf(r);
    float bw = 16.f * s * __builtin_sqrtf(1.f / r);
    float sx = (float)x * 16.f, sy = (float)y * 16.f;
    float ax1 = sx - 0.5f * bw, ay1 = sy - 0.5f * bh;
    float ax2 = sx + 0.5f * bw, ay2 = sy + 0.5f * bh;
    if (img == 0) {
        anchor_out[ai * 4 + 0] = ax1; anchor_out[ai * 4 + 1] = ay1;
        anchor_out[ai * 4 + 2] = ax2; anchor_out[ai * 4 + 3] = ay2;
    }

    const float* lp = out_locs + (long)img * 147456 + (long)site * 36 + a * 4;
    float dx = lp[0], dy = lp[1], dw = lp[2], dh = lp[3];
    float w = ax2 - ax1, h = ay2 - ay1;
    float cx = ax1 + 0.5f * w + dx * w;
    float cy = ay1 + 0.5f * h + dy * h;
    float pw = __expf(dw) * w, ph = __expf(dh) * h;
    float imw = (float)(*p_imgw), imh = (float)(*p_imgh);
    float x1 = fminf(fmaxf(cx - 0.5f * pw, 0.f), imw);
    float y1 = fminf(fmaxf(cy - 0.5f * ph, 0.f), imh);
    float x2 = fminf(fmaxf(cx + 0.5f * pw, 0.f), imw);
    float y2 = fminf(fmaxf(cy + 0.5f * ph, 0.f), imh);

    const float* sp = out_scores + (long)img * 73728 + (long)site * 18 + a * 2;
    float s0 = sp[0], s1 = sp[1];
    float mx = fmaxf(s0, s1);
    float e0 = __expf(s0 - mx), e1 = __expf(s1 - mx);
    float fg = e1 / (e0 + e1);

    bool valid = (x2 - x1 >= 16.f) && (y2 - y1 >= 16.f);
    float sc = valid ? fg : -__builtin_inff();

    float* bp = boxes + ((long)img * NANCH + ai) * 4;
    bp[0] = x1; bp[1] = y1; bp[2] = x2; bp[3] = y2;
    nscore[(long)img * NANCH + ai] = sc;
}

// ---------------- K4: sequential NMS (one block per image) ----------------
__global__ __launch_bounds__(1024) void k_nms(const float* __restrict__ boxes,
                                              const float* __restrict__ nscore,
                                              float* __restrict__ rois,
                                              float* __restrict__ roi_idx) {
    extern __shared__ float smem[];
    float* s  = smem;                         // NANCH scores
    float* rv = smem + NANCH;                 // 1024 reduce vals
    int*   ri = (int*)(smem + NANCH + 1024);  // 1024 reduce idxs

    const int n   = blockIdx.x;
    const int tid = threadIdx.x;
    const float* B = boxes + (long)n * NANCH * 4;
    const float NEG = -__builtin_inff();

    for (int i = tid; i < NANCH; i += 1024) s[i] = nscore[(long)n * NANCH + i];
    roi_idx[n * POSTN + tid] = (float)n;
    if (tid + 1024 < POSTN) roi_idx[n * POSTN + tid + 1024] = (float)n;
    __syncthreads();

    for (int t = 0; t < POSTN; ++t) {
        // local argmax (first-index tie break, like jnp.argmax)
        float bv = NEG; int bi = NANCH - 1;
        for (int i = tid; i < NANCH; i += 1024) {
            float v = s[i];
            if (v > bv || (v == bv && i < bi)) { bv = v; bi = i; }
        }
        rv[tid] = bv; ri[tid] = bi;
        __syncthreads();
        for (int off = 512; off > 0; off >>= 1) {
            if (tid < off) {
                float v2 = rv[tid + off]; int i2 = ri[tid + off];
                if (v2 > rv[tid] || (v2 == rv[tid] && i2 < ri[tid])) {
                    rv[tid] = v2; ri[tid] = i2;
                }
            }
            __syncthreads();
        }
        int   sel  = ri[0];
        float sval = rv[0];
        bool  valid = sval > NEG;
        float bx1 = B[sel * 4 + 0], by1 = B[sel * 4 + 1];
        float bx2 = B[sel * 4 + 2], by2 = B[sel * 4 + 3];
        float barea = (bx2 - bx1) * (by2 - by1);
        if (tid == 0) {
            float* rp = rois + ((long)n * POSTN + t) * 4;
            rp[0] = valid ? bx1 : 0.f; rp[1] = valid ? by1 : 0.f;
            rp[2] = valid ? bx2 : 0.f; rp[3] = valid ? by2 : 0.f;
        }
        // suppress IoU > 0.7 (and the selected box)
        for (int i = tid; i < NANCH; i += 1024) {
            float v = s[i];
            if (v == NEG) continue;
            float cx1 = B[i * 4 + 0], cy1 = B[i * 4 + 1];
            float cx2 = B[i * 4 + 2], cy2 = B[i * 4 + 3];
            float xx1 = fmaxf(bx1, cx1), yy1 = fmaxf(by1, cy1);
            float xx2 = fminf(bx2, cx2), yy2 = fminf(by2, cy2);
            float inter = fmaxf(xx2 - xx1, 0.f) * fmaxf(yy2 - yy1, 0.f);
            float area  = (cx2 - cx1) * (cy2 - cy1);
            float iou   = inter / (area + barea - inter + 1e-9f);
            if (iou > 0.7f || i == sel) s[i] = NEG;
        }
        __syncthreads();
    }
}

// ---------------- launcher ----------------
extern "C" void kernel_launch(void* const* d_in, const int* in_sizes, int n_in,
                              void* d_out, int out_size, void* d_ws, size_t ws_size,
                              hipStream_t stream) {
    const float* x       = (const float*)d_in[0];
    const float* conv1_w = (const float*)d_in[1];
    const float* conv1_b = (const float*)d_in[2];
    const float* score_w = (const float*)d_in[3];
    const float* score_b = (const float*)d_in[4];
    const float* loc_w   = (const float*)d_in[5];
    const float* loc_b   = (const float*)d_in[6];
    const int*   img_h   = (const int*)d_in[7];
    const int*   img_w   = (const int*)d_in[8];

    // d_out layout (floats): locs | scores | rois | roi_indices | anchor
    float* out       = (float*)d_out;
    float* o_locs    = out;                 // 8*36864*4 = 1,179,648
    float* o_scores  = out + 1179648;       // 8*36864*2 =   589,824
    float* o_rois    = out + 1769472;       // 16000*4   =    64,000
    float* o_ridx    = out + 1833472;       // 16000
    float* o_anchor  = out + 1849472;       // 36864*4   =   147,456

    // workspace layout
    char* ws = (char*)d_ws;
    _Float16* wf16  = (_Float16*)ws;                           //  4,718,592 B
    float*    h1    = (float*)(ws + 4718592);                  // 67,108,864 B
    float*    boxes = (float*)(ws + 4718592 + 67108864);       //  4,718,592 B
    float*    nsc   = (float*)(ws + 4718592 + 67108864 + 4718592);

    // K0: weight convert + permute (512*512*9 = 2,359,296 elements)
    k_cvt_w<<<9216, 256, 0, stream>>>(conv1_w, wf16, C_OUT * K_TOT);

    // K1: WMMA implicit-GEMM conv (512 M-tiles x 4 N-tiles)
    k_conv3<<<2048, 256, 0, stream>>>(x, wf16, conv1_b, h1);

    // K2: 1x1 heads (54 * 32768 threads)
    k_heads<<<6912, 256, 0, stream>>>(h1, score_w, score_b, loc_w, loc_b,
                                      o_scores, o_locs);

    // K3: decode (8 * 36864 threads)
    k_decode<<<1152, 256, 0, stream>>>(o_scores, o_locs, img_h, img_w,
                                       boxes, nsc, o_anchor);

    // K4: NMS, one block per image, scores in 152KB dynamic LDS
    size_t shmem = (size_t)(NANCH + 1024) * sizeof(float) + 1024 * sizeof(int);
    k_nms<<<N_IMG, 1024, shmem, stream>>>(boxes, nsc, o_rois, o_ridx);
}
